// GroupAttention_position_lepe_all_50714973831579
// MI455X (gfx1250) — matrix operand order repaired and checked by reading, not compiled, not run on hardware
//
#include <hip/hip_runtime.h>

// ---------------------------------------------------------------------------
// CDNA5 (gfx1250) wave32 implementation of windowed attention + LePE.
// - All matmuls on v_wmma_f32_16x16x32_bf16 (bf16 operands, fp32 accum).
// - qkv intermediate kept in bf16 (halves HBM traffic of the dominant term).
// - Q/V tiles staged into LDS by the Tensor Data Mover; K staged transposed
//   so S-stage B-fragment reads hit consecutive tokens across lanes.
// ---------------------------------------------------------------------------

typedef __bf16 v16bf __attribute__((ext_vector_type(16)));
typedef float  v8f   __attribute__((ext_vector_type(8)));
typedef unsigned int v4u __attribute__((ext_vector_type(4)));
typedef int          v4i __attribute__((ext_vector_type(4)));
typedef int          v8i __attribute__((ext_vector_type(8)));

#if __has_builtin(__builtin_amdgcn_tensor_load_to_lds) && \
    __has_builtin(__builtin_amdgcn_s_wait_tensorcnt)
#define USE_TDM 1
#else
#define USE_TDM 0
#endif

__device__ __forceinline__ unsigned pack_bf16(float lo, float hi) {
  unsigned short a = __builtin_bit_cast(unsigned short, (__bf16)lo);
  unsigned short b = __builtin_bit_cast(unsigned short, (__bf16)hi);
  return (unsigned)a | ((unsigned)b << 16);
}

// ---------------------------------------------------------------------------
// Generic GEMM: out[M,N] = A[M,K] @ W[N,K]^T (+ bias). 64x64x32 tiles,
// 128 threads = 4 waves, each wave computes a 16x64 strip (4 WMMA tiles).
// M%64==0, N%64==0, K%32==0 guaranteed by caller. OUT_BF16 selects the
// output element type (bf16 for the qkv intermediate, fp32 for the result).
// ---------------------------------------------------------------------------
template <bool OUT_BF16>
__global__ __launch_bounds__(128)
void gemm_xwT_bf16(const float* __restrict__ A, const float* __restrict__ W,
                   const float* __restrict__ bias, void* __restrict__ outv,
                   int M, int N, int K) {
  __shared__ __bf16 As[64][32];   // row-major M x K tile
  __shared__ __bf16 Bs[32][64];   // K x N tile (transposed from W)

  const int tid  = threadIdx.x;
  const int wave = tid >> 5;
  const int lane = tid & 31;
  const int half = lane >> 4;
  const int l16  = lane & 15;
  const int m0 = blockIdx.x * 64;
  const int n0 = blockIdx.y * 64;

  v8f acc[4] = {};

  for (int k0 = 0; k0 < K; k0 += 32) {
    // Stage A tile (64x32): consecutive tids read consecutive k -> coalesced.
#pragma unroll
    for (int t = 0; t < 16; ++t) {
      int idx = t * 128 + tid;
      int r = idx >> 5, c = idx & 31;
      As[r][c] = (__bf16)A[(long)(m0 + r) * K + k0 + c];
    }
    // Stage B tile transposed: Bs[k][n] = W[n0+n][k0+k].
    // Consecutive tids read consecutive k (contiguous in W) -> coalesced.
#pragma unroll
    for (int t = 0; t < 16; ++t) {
      int idx = t * 128 + tid;
      int kk = idx & 31, n = idx >> 5;
      Bs[kk][n] = (__bf16)W[(long)(n0 + n) * K + k0 + kk];
    }
    // Prefetch next K-slab of the streaming operands into cache.
    if (k0 + 32 < K) {
      __builtin_prefetch(&A[(long)(m0 + (tid >> 1)) * K + k0 + 32], 0, 3);
      __builtin_prefetch(&W[(long)(n0 + (tid >> 1)) * K + k0 + 32], 0, 3);
    }
    __syncthreads();

    // A fragment: 16-bit A 16x32 layout — lanes 0-15: M=lane, VGPR j pairs
    // K={2j,2j+1}(+8 for lanes 16-31); VGPRs 4-7 cover K=16..31.
    v16bf a;
#pragma unroll
    for (int j = 0; j < 8; ++j) {
      int kb = (j < 4) ? (half * 8 + 2 * j) : (16 + half * 8 + 2 * (j - 4));
      a[2 * j]     = As[wave * 16 + l16][kb];
      a[2 * j + 1] = As[wave * 16 + l16][kb + 1];
    }
#pragma unroll
    for (int nt = 0; nt < 4; ++nt) {
      // B fragment: N = lane%16, K in same order as the A fragment.
      v16bf b;
#pragma unroll
      for (int j = 0; j < 8; ++j) {
        int kb = (j < 4) ? (half * 8 + 2 * j) : (16 + half * 8 + 2 * (j - 4));
        b[2 * j]     = Bs[kb][nt * 16 + l16];
        b[2 * j + 1] = Bs[kb + 1][nt * 16 + l16];
      }
      acc[nt] = __builtin_amdgcn_wmma_f32_16x16x32_bf16(
          false, a, false, b, (short)0, acc[nt], false, false);
    }
    __syncthreads();
  }

  // C/D layout: VGPR r -> M = r + 8*(lane/16), N = lane%16.
#pragma unroll
  for (int nt = 0; nt < 4; ++nt) {
#pragma unroll
    for (int r = 0; r < 8; ++r) {
      int m = m0 + wave * 16 + r + 8 * half;
      int n = n0 + nt * 16 + l16;
      float v = acc[nt][r];
      if (bias) v += bias[n];
      if constexpr (OUT_BF16)
        ((__bf16*)outv)[(long)m * N + n] = (__bf16)v;
      else
        ((float*)outv)[(long)m * N + n] = v;
    }
  }
}

// ---------------------------------------------------------------------------
// Windowed attention: one block (128 thr / 4 waves) per (window, head).
// Q/V tiles (49x32 bf16, 3-D strided) DMA'd to LDS by the Tensor Data Mover;
// K staged transposed (Kt[d][token]) so S-stage B fragments are contiguous.
// 49 tokens padded to 64 with zeros.
// ---------------------------------------------------------------------------
__global__ __launch_bounds__(128)
void win_attn(const __bf16* __restrict__ qkv, float* __restrict__ out) {
  constexpr int WSS = 49;
  __shared__ __bf16 Qs[64][32];   // row-major (token, channel)
  __shared__ __bf16 Vs[64][32];   // row-major (token, channel)
  __shared__ __bf16 Kt[32][64];   // transposed (channel, token)
  __shared__ float  St[64][64];   // S^T: St[j][i] = S[i][j]
  __shared__ __bf16 Ps[64][64];   // P row-major (token, token)

  const int tid  = threadIdx.x;
  const int wave = tid >> 5;
  const int lane = tid & 31;
  const int half = lane >> 4;
  const int l16  = lane & 15;
  const int wwin = blockIdx.x;          // 0..63 (8x8 window grid)
  const int head = blockIdx.y;          // 0..11
  const int b    = blockIdx.z;          // 0..7
  const int wy = wwin >> 3, wx = wwin & 7;
  const long tok0 = (long)b * 3136 + (wy * 7) * 56 + wx * 7;

#if USE_TDM
  if (wave == 0) {
    // Tensor DMA descriptor (CDNA5 ISA):
    // tile = 32 contiguous bf16 channels x 7 tokens (stride 1152 elems)
    //        x 7 rows (stride 56*1152 elems), dest = contiguous LDS rows.
    v8i g1;
    g1[0] = (1 << 16);                 // workgroup_mask=0, data_size=1 (2B)
    g1[1] = (32 << 16);                // tensor_dim0[15:0]=32 @ bits 63:48
    g1[2] = (56 << 16);                // dim0 hi=0 | tensor_dim1[15:0]=56
    g1[3] = (32 << 16);                // dim1 hi=0 | tile_dim0=32
    g1[4] = 7 | (7 << 16);             // tile_dim1=7, tile_dim2=7
    g1[5] = 1152;                      // tensor_dim0_stride[31:0]
    g1[6] = (int)((64512u & 0xffffu) << 16);  // stride0 hi=0 | stride1 lo
    g1[7] = (int)(64512u >> 16);       // tensor_dim1_stride[47:16]
    v4i g2;
    g2[0] = 56;                        // tensor_dim2 (OOB bound for Z)
    g2[1] = 0; g2[2] = 0; g2[3] = 0;   // no dim3 / iteration
    v4i g3 = {0, 0, 0, 0};
    v8i g4 = {0, 0, 0, 0, 0, 0, 0, 0}; // unused trailing group (6-arg form)

    const unsigned lds_base[2] = {
        (unsigned)(unsigned long long)&Qs[0][0],
        (unsigned)(unsigned long long)&Vs[0][0]};
    const int      matoff[2] = {0, 768};
    const unsigned long long ga0 =
        (unsigned long long)(qkv + tok0 * 1152 + head * 32);
#pragma unroll
    for (int m = 0; m < 2; ++m) {
      unsigned long long ga = ga0 + (unsigned long long)matoff[m] * 2;
      v4u g0;
      g0[0] = 1u;                                  // count=1 (valid), user mode
      g0[1] = lds_base[m];                         // lds_addr
      g0[2] = (unsigned)(ga & 0xffffffffu);        // global_addr[31:0]
      g0[3] = (unsigned)((ga >> 32) & 0x1ffffffu)  // global_addr[56:32]
              | (2u << 30);                        // type=2 ("image")
      __builtin_amdgcn_tensor_load_to_lds(g0, g1, g2, g3, g4, 0);
    }
  }
#else
  for (int idx = tid; idx < WSS * 32; idx += 128) {
    int r = idx >> 5, d = idx & 31;
    int y = wy * 7 + r / 7, x = wx * 7 + r % 7;
    long base = ((long)b * 3136 + y * 56 + x) * 1152 + head * 32 + d;
    Qs[r][d] = qkv[base];
    Vs[r][d] = qkv[base + 768];
  }
#endif

  // Zero pad rows (49..63) of Q/V while the TDM streams rows 0..48.
  for (int idx = tid; idx < 15 * 32; idx += 128) {
    int r = 49 + (idx >> 5), d = idx & 31;
    Qs[r][d] = (__bf16)0.f;
    Vs[r][d] = (__bf16)0.f;
  }

  // Stage K transposed: Kt[d][j] = K[token j][channel d] (zeros for j>=49).
  // Consecutive tids read consecutive channels -> coalesced global loads.
  for (int idx = tid; idx < 64 * 32; idx += 128) {
    int j = idx >> 5, d = idx & 31;
    __bf16 kv = (__bf16)0.f;
    if (j < WSS) {
      int y = wy * 7 + j / 7, x = wx * 7 + j % 7;
      kv = qkv[((long)b * 3136 + y * 56 + x) * 1152 + 384 + head * 32 + d];
    }
    Kt[d][j] = kv;
  }

#if USE_TDM
  __builtin_amdgcn_s_wait_tensorcnt(0);   // no-op for waves 1..3
#endif
  __syncthreads();

  // S = Q K^T : per wave 16 rows x 64 cols, K-dim = 32 (one WMMA step).
  {
    v16bf a;
#pragma unroll
    for (int j = 0; j < 8; ++j) {
      int kb = (j < 4) ? (half * 8 + 2 * j) : (16 + half * 8 + 2 * (j - 4));
      a[2 * j]     = Qs[wave * 16 + l16][kb];
      a[2 * j + 1] = Qs[wave * 16 + l16][kb + 1];
    }
#pragma unroll
    for (int nt = 0; nt < 4; ++nt) {
      v16bf bm;   // B[k=d][n=j] = Kt[d][j]: n = lane%16, k in A order
#pragma unroll
      for (int j = 0; j < 8; ++j) {
        int kb = (j < 4) ? (half * 8 + 2 * j) : (16 + half * 8 + 2 * (j - 4));
        bm[2 * j]     = Kt[kb][nt * 16 + l16];
        bm[2 * j + 1] = Kt[kb + 1][nt * 16 + l16];
      }
      v8f s = {};
      s = __builtin_amdgcn_wmma_f32_16x16x32_bf16(
          false, a, false, bm, (short)0, s, false, false);
      // Store S transposed: 8 consecutive floats per lane -> ds_store_b128.
#pragma unroll
      for (int r = 0; r < 8; ++r)
        St[nt * 16 + l16][wave * 16 + 8 * half + r] = s[r];
    }
  }
  __syncthreads();

  // Row softmax over the 49 valid columns (S row i = St[:,i]); emit bf16 P.
  if (tid < 64) {
    int r = tid;
    unsigned* prow = (unsigned*)&Ps[r][0];
    if (r < WSS) {
      const float scale = 0.17677669529663687f;   // 32^-0.5
      float mx = -1e30f;
      for (int j = 0; j < WSS; ++j) mx = fmaxf(mx, St[j][r] * scale);
      float sum = 0.f;
      for (int j = 0; j < WSS; ++j) {
        float e = __expf(St[j][r] * scale - mx);
        St[j][r] = e;
        sum += e;
      }
      float inv = 1.f / sum;
#pragma unroll 4
      for (int jp = 0; jp < 32; ++jp) {
        int j0 = 2 * jp, j1 = 2 * jp + 1;
        float p0 = (j0 < WSS) ? St[j0][r] * inv : 0.f;
        float p1 = (j1 < WSS) ? St[j1][r] * inv : 0.f;
        prow[jp] = pack_bf16(p0, p1);
      }
    } else {
      for (int jp = 0; jp < 32; ++jp) prow[jp] = 0u;
    }
  }
  __syncthreads();

  // O = P @ V : K-dim = 64 (2 steps), N = 32 (2 tiles)
  v8f o[2] = {};
#pragma unroll
  for (int ks = 0; ks < 2; ++ks) {
    v16bf a;
#pragma unroll
    for (int j = 0; j < 8; ++j) {
      int kb = ((j < 4) ? (half * 8 + 2 * j) : (16 + half * 8 + 2 * (j - 4)))
               + ks * 32;
      a[2 * j]     = Ps[wave * 16 + l16][kb];
      a[2 * j + 1] = Ps[wave * 16 + l16][kb + 1];
    }
#pragma unroll
    for (int nt = 0; nt < 2; ++nt) {
      v16bf bm;   // B[k=j][n=d] = V[j][d]: n = lane%16, k in A order
#pragma unroll
      for (int j = 0; j < 8; ++j) {
        int kb = (j < 4) ? (half * 8 + 2 * j) : (16 + half * 8 + 2 * (j - 4));
        bm[2 * j]     = Vs[ks * 32 + kb][nt * 16 + l16];
        bm[2 * j + 1] = Vs[ks * 32 + kb + 1][nt * 16 + l16];
      }
      o[nt] = __builtin_amdgcn_wmma_f32_16x16x32_bf16(
          false, a, false, bm, (short)0, o[nt], false, false);
    }
  }

  // Scatter to image/token order (win2img fused into the store)
#pragma unroll
  for (int nt = 0; nt < 2; ++nt) {
#pragma unroll
    for (int r = 0; r < 8; ++r) {
      int i = wave * 16 + r + 8 * half;
      if (i < WSS) {
        int y = wy * 7 + i / 7, x = wx * 7 + i % 7;
        long oi = ((long)b * 3136 + y * 56 + x) * 384 + head * 32 + nt * 16 + l16;
        out[oi] = o[nt][r];
      }
    }
  }
}

// ---------------------------------------------------------------------------
// LePE: depthwise 3x3 conv over v (bf16, token order inside qkv buffer)
// + bias, accumulated into the attention output.
// ---------------------------------------------------------------------------
__global__ __launch_bounds__(256)
void lepe_add(const __bf16* __restrict__ qkv, const float* __restrict__ w,
              const float* __restrict__ bias, float* __restrict__ out,
              int total) {
  int idx = blockIdx.x * blockDim.x + threadIdx.x;
  if (idx >= total) return;
  int c = idx % 384;
  int x = (idx / 384) % 56;
  int y = (idx / (384 * 56)) % 56;
  int b = idx / (384 * 56 * 56);
  float acc = bias[c];
#pragma unroll
  for (int ky = 0; ky < 3; ++ky) {
    int yy = y + ky - 1;
    if (yy < 0 || yy >= 56) continue;
#pragma unroll
    for (int kx = 0; kx < 3; ++kx) {
      int xx = x + kx - 1;
      if (xx < 0 || xx >= 56) continue;
      long vi = ((long)b * 3136 + yy * 56 + xx) * 1152 + 768 + c;
      acc += (float)qkv[vi] * w[c * 9 + ky * 3 + kx];
    }
  }
  out[idx] += acc;
}

// ---------------------------------------------------------------------------
extern "C" void kernel_launch(void* const* d_in, const int* in_sizes, int n_in,
                              void* d_out, int out_size, void* d_ws, size_t ws_size,
                              hipStream_t stream) {
  const float* x      = (const float*)d_in[0];   // (8, 3136, 384)
  const float* qkv_w  = (const float*)d_in[1];   // (1152, 384)
  const float* proj_w = (const float*)d_in[2];   // (384, 384)
  const float* proj_b = (const float*)d_in[3];   // (384,)
  const float* lepe_w = (const float*)d_in[4];   // (384,1,3,3)
  const float* lepe_b = (const float*)d_in[5];   // (384,)
  float* outp = (float*)d_out;                   // (8, 3136, 384)

  __bf16* qkv = (__bf16*)d_ws;                               // 8*3136*1152 bf16
  float*  attn = (float*)((char*)d_ws +
                          (size_t)8 * 3136 * 1152 * sizeof(__bf16));
                                                             // 8*3136*384 fp32

  const int Mtok = 8 * 3136;   // 25088 (divisible by 64)

  // 1) qkv = x @ qkv_w^T  -> bf16   (25088 x 1152, K=384)
  gemm_xwT_bf16<true><<<dim3(Mtok / 64, 1152 / 64), 128, 0, stream>>>(
      x, qkv_w, nullptr, qkv, Mtok, 1152, 384);

  // 2) windowed attention over 7x7 windows, 12 heads, 8 batches
  win_attn<<<dim3(64, 12, 8), 128, 0, stream>>>(qkv, attn);

  // 3) LePE depthwise conv + bias, accumulated into attn
  int total = Mtok * 384;
  lepe_add<<<(total + 255) / 256, 256, 0, stream>>>(
      qkv, lepe_w, lepe_b, attn, total);

  // 4) out = attn @ proj_w^T + proj_b  -> fp32  (25088 x 384, K=384)
  gemm_xwT_bf16<false><<<dim3(Mtok / 64, 384 / 64), 128, 0, stream>>>(
      attn, proj_w, proj_b, outp, Mtok, 384, 384);
}
